// update_e_17377437680123
// MI455X (gfx1250) — matrix operand order repaired and checked by reading, not compiled, not run on hardware
//
#include <hip/hip_runtime.h>

// ---------------------------------------------------------------------------
// update_e for SchNet-style interaction block, MI455X (gfx1250, wave32).
//  e = (v @ Wlin)[j] * ( ssp(dist_emb @ W1 + b1) @ W2 + b2 ) * Cenv(dist)
// GEMMs on bf16 WMMA 16x16x32 (f32 accumulate). 32-row M-tiles per block:
// each wave runs two A tiles per B fragment (2x B reuse). A staged once per
// block in LDS (bf16, conflict-free pitches); B from pre-packed bf16 weights.
// ---------------------------------------------------------------------------

typedef __attribute__((ext_vector_type(16))) __bf16  v16bf;
typedef __attribute__((ext_vector_type(8)))  __bf16  v8bf;
typedef __attribute__((ext_vector_type(4)))  __bf16  v4bf;
typedef __attribute__((ext_vector_type(8)))  float   v8f;
typedef __attribute__((ext_vector_type(4)))  float   f4;

#define HIDDEN    128
#define NFILT     128
#define NGAUSS    50
#define SSP_SHIFT 0.6931471805599453f
#define ENV_W     0.31415926535897932f   // pi / cutoff(10)

// packed-weight region sizes (bf16 elements), layout pk[(kb*128+n)*32 + t]
#define P_W1  (2 * NFILT * 32)   //  8192  (K 50 -> 64, zero padded)
#define P_W2  (4 * NFILT * 32)   // 16384
#define P_WL  (4 * NFILT * 32)   // 16384
#define P_TOT (P_W1 + P_W2 + P_WL)

// LDS row pitches (bf16 elements): multiples of 8 (16B ds_load_b128 align),
// chosen so lane m's 4-dword read starts at bank (pitch/2*m) mod 64, all
// distinct, stride 4 -> conflict-free b128 reads.
#define PITCH_A  72     // 64-col A tile (edge kernel)
#define PITCH_H  136    // 128-col tiles (h tile / v tile)

__device__ __forceinline__ v16bf cat8(v8bf a, v8bf b) {
  return __builtin_shufflevector(a, b, 0,1,2,3,4,5,6,7,8,9,10,11,12,13,14,15);
}

__device__ __forceinline__ v8f bfwmma(v16bf a, v16bf b, v8f c) {
  return __builtin_amdgcn_wmma_f32_16x16x32_bf16(false, a, false, b,
                                                 (short)0, c, false, false);
}

// stable shifted softplus: max(x,0) + log(1+exp(-|x|)) - ln2
__device__ __forceinline__ float sspf(float x) {
  float t = __expf(-fabsf(x));
  return fmaxf(x, 0.0f) + __logf(1.0f + t) - SSP_SHIFT;
}

// ---------------------------------------------------------------------------
// Kernel 0: round weights to bf16 in WMMA-B layout: pk[(kb*128+n)*32 + t]
//           holds W[kb*32+t][n]; W1 zero-padded for k >= 50.
// ---------------------------------------------------------------------------
__global__ void pack_weights_kernel(const float* __restrict__ W1,
                                    const float* __restrict__ W2,
                                    const float* __restrict__ WL,
                                    __bf16* __restrict__ pk) {
  int idx = blockIdx.x * blockDim.x + threadIdx.x;
  if (idx >= P_TOT) return;
  float val;
  if (idx < P_W1) {
    int kb = idx / (NFILT * 32);
    int n  = (idx / 32) % NFILT;
    int t  = idx & 31;
    int k  = kb * 32 + t;
    val = (k < NGAUSS) ? W1[k * NFILT + n] : 0.0f;
  } else if (idx < P_W1 + P_W2) {
    int l  = idx - P_W1;
    int kb = l / (NFILT * 32);
    int n  = (l / 32) % NFILT;
    int t  = l & 31;
    val = W2[(kb * 32 + t) * NFILT + n];
  } else {
    int l  = idx - (P_W1 + P_W2);
    int kb = l / (NFILT * 32);
    int n  = (l / 32) % NFILT;
    int t  = l & 31;
    val = WL[(kb * 32 + t) * NFILT + n];
  }
  pk[idx] = (__bf16)val;
}

// ---------------------------------------------------------------------------
// Kernel 1: vproj = v @ Wlin   [NN,128] x [128,128] -> f32 workspace
// Block = 32 rows, 8 waves; per wave: 2 M-tiles x 4 K-steps = 8 WMMAs,
// B fragment loaded once per K-step.
// ---------------------------------------------------------------------------
__global__ void gemm_vproj_kernel(const float* __restrict__ v,
                                  const __bf16* __restrict__ packWL,
                                  float* __restrict__ vproj, int NN) {
  __shared__ __attribute__((aligned(16))) __bf16 vl[32][PITCH_H];

  const int t = threadIdx.x;
  {  // cooperative stage: thread t -> row t/8, 16 consecutive k
    const int r  = t >> 3;
    const int c0 = (t & 7) * 16;
    int row  = blockIdx.x * 32 + r;
    int rrow = (row < NN) ? row : (NN - 1);
    const float* arow = v + (size_t)rrow * HIDDEN;
    f4 x0 = *(const f4*)(arow + c0);
    f4 x1 = *(const f4*)(arow + c0 + 4);
    f4 x2 = *(const f4*)(arow + c0 + 8);
    f4 x3 = *(const f4*)(arow + c0 + 12);
    *(v4bf*)&vl[r][c0]      = __builtin_convertvector(x0, v4bf);
    *(v4bf*)&vl[r][c0 + 4]  = __builtin_convertvector(x1, v4bf);
    *(v4bf*)&vl[r][c0 + 8]  = __builtin_convertvector(x2, v4bf);
    *(v4bf*)&vl[r][c0 + 12] = __builtin_convertvector(x3, v4bf);
  }
  __syncthreads();

  const int wave  = t >> 5;
  const int lane  = t & 31;
  const int nb    = wave * 16 + (lane & 15);  // output column (N)
  const int hi    = lane >> 4;                // half-wave id
  const int klo   = hi * 8;                   // A-fragment K base
  const int khalf = hi * 16;                  // unused; B fragment also uses klo
  const int m     = lane & 15;                // A row within tile

  v8f acc0 = {}, acc1 = {};
#pragma unroll
  for (int kb = 0; kb < 4; ++kb) {
    const int k0 = kb * 32;
    const v8bf* bp = (const v8bf*)(packWL + ((size_t)(kb * NFILT + nb) * 32 + klo));
    v16bf b  = cat8(bp[0], bp[2]);
    v16bf a0 = cat8(*(const v8bf*)&vl[m][k0 + klo],
                    *(const v8bf*)&vl[m][k0 + klo + 16]);
    v16bf a1 = cat8(*(const v8bf*)&vl[m + 16][k0 + klo],
                    *(const v8bf*)&vl[m + 16][k0 + klo + 16]);
    acc0 = bfwmma(a0, b, acc0);
    acc1 = bfwmma(a1, b, acc1);
  }

  const int row0 = blockIdx.x * 32 + 8 * hi;  // D layout: reg r -> row0 + r
  if (blockIdx.x * 32 + 32 <= NN) {           // uniform fast path
    float* p = vproj + (size_t)row0 * NFILT + nb;
#pragma unroll
    for (int r = 0; r < 8; ++r) {
      p[r * NFILT]        = acc0[r];
      p[(16 + r) * NFILT] = acc1[r];
    }
  } else {                                    // tail block
#pragma unroll
    for (int r = 0; r < 8; ++r) {
      int M0 = row0 + r, M1 = row0 + 16 + r;
      if (M0 < NN) vproj[(size_t)M0 * NFILT + nb] = acc0[r];
      if (M1 < NN) vproj[(size_t)M1 * NFILT + nb] = acc1[r];
    }
  }
}

// ---------------------------------------------------------------------------
// Kernel 2: per-edge fused filter MLP + envelope + gather + multiply.
// Block = 32 edges, 8 waves; wave w owns filter columns [16w,16w+16),
// two edge tiles per wave (12 WMMAs/wave).
// ---------------------------------------------------------------------------
__global__ void edge_kernel(const float* __restrict__ dist,
                            const float* __restrict__ demb,
                            const int*  __restrict__ jidx,
                            const float* __restrict__ b1,
                            const float* __restrict__ b2,
                            const __bf16* __restrict__ packW1,
                            const __bf16* __restrict__ packW2,
                            const float* __restrict__ vproj,
                            float* __restrict__ out) {
  __shared__ __attribute__((aligned(16))) __bf16 al[32][PITCH_A];  // dist_emb tile
  __shared__ __attribute__((aligned(16))) __bf16 hl[32][PITCH_H];  // h tile
  __shared__ float cl[32];   // cosine envelope per edge
  __shared__ int   jl[32];   // source node per edge

  const int e0 = blockIdx.x * 32;
  const int t  = threadIdx.x;

  {  // cooperative stage: 32 x 50 f32 -> 32 x 64 bf16, zero padded.
     // Loads are unconditional (index clamped in-row), tail zeroed by select.
    const int r  = t >> 3;
    const int c0 = (t & 7) * 8;
    const float* drow = demb + (size_t)(e0 + r) * NGAUSS;
    f4 x0, x1;
#pragma unroll
    for (int i = 0; i < 4; ++i) {
      int k = c0 + i;
      float xv = drow[(k < NGAUSS) ? k : (NGAUSS - 1)];
      x0[i] = (k < NGAUSS) ? xv : 0.0f;
    }
#pragma unroll
    for (int i = 0; i < 4; ++i) {
      int k = c0 + 4 + i;
      float xv = drow[(k < NGAUSS) ? k : (NGAUSS - 1)];
      x1[i] = (k < NGAUSS) ? xv : 0.0f;
    }
    *(v4bf*)&al[r][c0]     = __builtin_convertvector(x0, v4bf);
    *(v4bf*)&al[r][c0 + 4] = __builtin_convertvector(x1, v4bf);
  }
  if (t < 32) {
    float d = dist[e0 + t];
    cl[t] = 0.5f * (__cosf(d * ENV_W) + 1.0f);
    jl[t] = jidx[e0 + t];
  }
  __syncthreads();

  const int wave  = t >> 5;
  const int lane  = t & 31;
  const int nb    = wave * 16 + (lane & 15);
  const int hi    = lane >> 4;
  const int klo   = hi * 8;
  const int khalf = hi * 16;
  const int m     = lane & 15;

  // ---- GEMM 1: h = ssp(dist_emb @ W1 + b1), K = 50 padded to 64 ----
  v8f h0 = {}, h1 = {};
#pragma unroll
  for (int kb = 0; kb < 2; ++kb) {
    const int k0 = kb * 32;
    const v8bf* bp = (const v8bf*)(packW1 + ((size_t)(kb * NFILT + nb) * 32 + klo));
    v16bf b  = cat8(bp[0], bp[2]);
    v16bf a0 = cat8(*(const v8bf*)&al[m][k0 + klo],
                    *(const v8bf*)&al[m][k0 + klo + 16]);
    v16bf a1 = cat8(*(const v8bf*)&al[m + 16][k0 + klo],
                    *(const v8bf*)&al[m + 16][k0 + klo + 16]);
    h0 = bfwmma(a0, b, h0);
    h1 = bfwmma(a1, b, h1);
  }

  // bias + shifted softplus; stage h tile in LDS as bf16 (rows=edges)
  const float b1c = b1[nb];
#pragma unroll
  for (int r = 0; r < 8; ++r) {
    hl[r + 8 * hi][nb]      = (__bf16)sspf(h0[r] + b1c);
    hl[16 + r + 8 * hi][nb] = (__bf16)sspf(h1[r] + b1c);
  }
  __syncthreads();

  // ---- GEMM 2: W = h @ W2, K = 128 = 4 steps; A straight from LDS ----
  v8f w0 = {}, w1 = {};
#pragma unroll
  for (int kb = 0; kb < 4; ++kb) {
    const int k0 = kb * 32;
    const v8bf* bp = (const v8bf*)(packW2 + ((size_t)(kb * NFILT + nb) * 32 + klo));
    v16bf b  = cat8(bp[0], bp[2]);
    v16bf a0 = cat8(*(const v8bf*)&hl[m][k0 + klo],
                    *(const v8bf*)&hl[m][k0 + klo + 16]);
    v16bf a1 = cat8(*(const v8bf*)&hl[m + 16][k0 + klo],
                    *(const v8bf*)&hl[m + 16][k0 + klo + 16]);
    w0 = bfwmma(a0, b, w0);
    w1 = bfwmma(a1, b, w1);
  }

  // ---- bias + envelope + gather vproj[j] (L2-resident table) + store ----
  const float b2c = b2[nb];
  float* op = out + (size_t)(e0 + 8 * hi) * NFILT + nb;  // reg r -> +r rows
#pragma unroll
  for (int r = 0; r < 8; ++r) {
    const int M0 = r + 8 * hi;
    const int M1 = M0 + 16;
    const float wv0 = (w0[r] + b2c) * cl[M0];
    const float wv1 = (w1[r] + b2c) * cl[M1];
    op[r * NFILT]        = vproj[(size_t)jl[M0] * NFILT + nb] * wv0;
    op[(16 + r) * NFILT] = vproj[(size_t)jl[M1] * NFILT + nb] * wv1;
  }
}

// ---------------------------------------------------------------------------
extern "C" void kernel_launch(void* const* d_in, const int* in_sizes, int n_in,
                              void* d_out, int out_size, void* d_ws, size_t ws_size,
                              hipStream_t stream) {
  const float* v    = (const float*)d_in[0];
  const float* dist = (const float*)d_in[1];
  const float* demb = (const float*)d_in[2];
  const int*   eidx = (const int*)d_in[3];   // edge_index[0] = j (first E entries)
  const float* W1   = (const float*)d_in[4];
  const float* b1   = (const float*)d_in[5];
  const float* W2   = (const float*)d_in[6];
  const float* b2   = (const float*)d_in[7];
  const float* Wlin = (const float*)d_in[8];
  float*       out  = (float*)d_out;

  const int E  = in_sizes[1];            // 1,600,000  (divisible by 32)
  const int NN = in_sizes[0] / HIDDEN;   // 50,000

  // workspace: [ vproj f32 NN*128 | packed bf16 weights ]
  float*  vproj = (float*)d_ws;
  __bf16* pk    = (__bf16*)(vproj + (size_t)NN * HIDDEN);
  __bf16* pW1   = pk;
  __bf16* pW2   = pk + P_W1;
  __bf16* pWL   = pk + P_W1 + P_W2;

  pack_weights_kernel<<<(P_TOT + 255) / 256, 256, 0, stream>>>(W1, W2, Wlin, pk);
  gemm_vproj_kernel<<<(NN + 31) / 32, 256, 0, stream>>>(v, pWL, vproj, NN);
  edge_kernel<<<E / 32, 256, 0, stream>>>(dist, demb, eidx, b1, b2, pW1, pW2,
                                          vproj, out);
}
